// DecodeBlock_71262097375684
// MI455X (gfx1250) — matrix operand order repaired and checked by reading, not compiled, not run on hardware
//
#include <hip/hip_runtime.h>

#define BB 8
#define SS 1024
#define EE 1024
#define HH 8
#define DD 128
#define MM (BB*SS)
#define GN 1024   // GEMM N (compile-time)
#define GK 1024   // GEMM K (compile-time)

typedef unsigned short u16;
typedef __bf16 bf16x16 __attribute__((ext_vector_type(16)));
typedef float  floatx8 __attribute__((ext_vector_type(8)));

union FragU { bf16x16 v; uint4 q[2]; };

__device__ __forceinline__ u16 f2bf(float f){
  unsigned u = __float_as_uint(f);
  u += 0x7FFFu + ((u >> 16) & 1u);          // round-to-nearest-even
  return (u16)(u >> 16);
}
__device__ __forceinline__ float bf2f(u16 h){
  return __uint_as_float(((unsigned)h) << 16);
}

// A fragment (16x32 bf16, M x K): lane<16 -> M=lane, K 0..7 & 16..23; lane>=16 -> K 8..15 & 24..31
__device__ __forceinline__ bf16x16 load_a16(const u16* base, int ld, int row0, int k0, int lane){
  int m = lane & 15; int ks = (lane < 16) ? 0 : 8;
  const u16* p = base + (size_t)(row0 + m) * ld + k0 + ks;
  FragU f;
  f.q[0] = *(const uint4*)(p);
  f.q[1] = *(const uint4*)(p + 16);
  return f.v;
}
// B fragment (32x16 bf16, K x N) from a row-major (N x K) matrix: lane<16 -> N=lane, K 0..7 & 16..23; lane>=16 -> K 8..15 & 24..31
__device__ __forceinline__ bf16x16 load_b16(const u16* base, int ld, int row0, int k0, int lane){
  int n = lane & 15; int ks = (lane < 16) ? 0 : 8;
  const u16* p = base + (size_t)(row0 + n) * ld + k0 + ks;
  FragU f;
  f.q[0] = *(const uint4*)(p);
  f.q[1] = *(const uint4*)(p + 16);
  return f.v;
}
__device__ __forceinline__ floatx8 wmma_bf(bf16x16 a, bf16x16 b, floatx8 c){
  return __builtin_amdgcn_wmma_f32_16x16x32_bf16(false, a, false, b, (short)0, c, false, false);
}

// ---------------- conversion / utility kernels ----------------

__global__ void k_cvt(const float* __restrict__ src, u16* __restrict__ dst, size_t n){
  size_t i = (size_t)blockIdx.x * blockDim.x + threadIdx.x;
  if (i < n) dst[i] = f2bf(src[i]);
}

// batched square transpose + convert: dst[z][i][j] = bf16(src[z][j][i]), n x n
__global__ void k_trans(const float* __restrict__ src, u16* __restrict__ dst, int n){
  int j = blockIdx.x * 16 + threadIdx.x;   // dst col (contiguous writes)
  int i = blockIdx.y * 16 + threadIdx.y;   // dst row
  int z = blockIdx.z;
  dst[((size_t)z * n + i) * n + j] = f2bf(src[((size_t)z * n + j) * n + i]);
}

__global__ void k_seg(const unsigned char* __restrict__ dones, int* __restrict__ c){
  int b = threadIdx.x;
  if (b >= BB) return;
  int acc = 0;
  for (int s = 0; s < SS; ++s){
    acc += dones[b * SS + s] ? 1 : 0;
    c[b * SS + s] = acc;
  }
}

// ---------------- GEMM: C[M,1024] = A[M,1024] @ WT[1024,1024]^T  (bf16 in, f32 out) ----------------
// Tile staging uses CDNA5 async memory->LDS copies (ASYNCcnt) with double buffering so the
// HBM/L2 fetch of K-step i+1 overlaps the WMMAs of K-step i.

__global__ __launch_bounds__(256) void k_gemm(const u16* __restrict__ A, const u16* __restrict__ WT,
                                              float* __restrict__ C){
  __shared__ alignas(16) u16 As[2][128*32];
  __shared__ alignas(16) u16 Bs[2][128*32];
  int tid = threadIdx.x, lane = tid & 31, w = tid >> 5;
  int m0 = blockIdx.y * 128, n0 = blockIdx.x * 128;
  int wm = (w >> 1) * 32, wn = (w & 1) * 64;
  floatx8 zero = {0.f,0.f,0.f,0.f,0.f,0.f,0.f,0.f};
  floatx8 acc[2][4];
  for (int i=0;i<2;i++) for (int j=0;j<4;j++) acc[i][j] = zero;

  int lrow = tid >> 1, lcol = (tid & 1) * 16;   // each thread owns 32B of a 128x32 tile
  unsigned ldsA[2], ldsB[2];
  ldsA[0] = (unsigned)(unsigned long long)(&As[0][lrow*32 + lcol]);
  ldsA[1] = (unsigned)(unsigned long long)(&As[1][lrow*32 + lcol]);
  ldsB[0] = (unsigned)(unsigned long long)(&Bs[0][lrow*32 + lcol]);
  ldsB[1] = (unsigned)(unsigned long long)(&Bs[1][lrow*32 + lcol]);
  unsigned long long Ap = (unsigned long long)A;
  unsigned long long Bp = (unsigned long long)WT;
  unsigned gaBase = (unsigned)((m0 + lrow) * GK + lcol) * 2u;   // byte offset
  unsigned gbBase = (unsigned)((n0 + lrow) * GK + lcol) * 2u;

  auto issue = [&](int k0, int sel){
    unsigned la = ldsA[sel], lb = ldsB[sel];
    unsigned ga = gaBase + (unsigned)k0 * 2u;
    unsigned gb = gbBase + (unsigned)k0 * 2u;
    // async copy: 16B from Ap/Bp + ga/gb into LDS at la/lb, then the next 16B via offset:16
    asm volatile("global_load_async_to_lds_b128 %0, %1, %2\n\t"
                 "global_load_async_to_lds_b128 %0, %1, %2 offset:16"
                 :: "v"(la), "v"(ga), "s"(Ap) : "memory");
    asm volatile("global_load_async_to_lds_b128 %0, %1, %2\n\t"
                 "global_load_async_to_lds_b128 %0, %1, %2 offset:16"
                 :: "v"(lb), "v"(gb), "s"(Bp) : "memory");
  };

  issue(0, 0);
  for (int k0 = 0; k0 < GK; k0 += 32){
    int cur = (k0 >> 5) & 1;
    if (k0 + 32 < GK){
      issue(k0 + 32, cur ^ 1);
      asm volatile("s_wait_asynccnt 0x4" ::: "memory");  // 4 in flight for next buffer; current done
    } else {
      asm volatile("s_wait_asynccnt 0x0" ::: "memory");
    }
    __syncthreads();
    const u16* Asc = &As[cur][0];
    const u16* Bsc = &Bs[cur][0];
    bf16x16 af[2], bfr[4];
    af[0] = load_a16(Asc, 32, wm,     0, lane);
    af[1] = load_a16(Asc, 32, wm+16,  0, lane);
    #pragma unroll
    for (int j=0;j<4;j++) bfr[j] = load_b16(Bsc, 32, wn + j*16, 0, lane);
    #pragma unroll
    for (int i=0;i<2;i++)
      #pragma unroll
      for (int j=0;j<4;j++) acc[i][j] = wmma_bf(af[i], bfr[j], acc[i][j]);
    __syncthreads();
  }
  int cn = lane & 15, chi = (lane < 16) ? 0 : 8;
  float* Cp = C + (size_t)(m0 + wm + chi) * GN + (n0 + wn + cn);
  #pragma unroll
  for (int i=0;i<2;i++)
    #pragma unroll
    for (int j=0;j<4;j++)
      #pragma unroll
      for (int r=0;r<8;r++)
        Cp[(size_t)(i*16 + r) * GN + j*16] = acc[i][j][r];
}

// ---------------- QKV post: split heads, scale k, rope q/k, write bf16 (v transposed) ----------------

__global__ __launch_bounds__(512) void k_qkvpost(const float* __restrict__ qf, const float* __restrict__ kf,
                                                 const float* __restrict__ vf, const int* __restrict__ pos,
                                                 u16* __restrict__ qbh, u16* __restrict__ kbh, u16* __restrict__ vT){
  int bs = blockIdx.x;               // b*SS + s
  int t = threadIdx.x;
  int h = t >> 6, j = t & 63;
  int b = bs >> 10, s = bs & 1023;
  int bh = b * HH + h;
  float p = (float)pos[bs];
  float freq = exp2f(-(float)j * (13.287712379549449f / 64.f));   // 10000^(-j/64)
  float ang = p * freq;
  float sn, cs;
  __sincosf(ang, &sn, &cs);
  const float SC = 0.08838834764831845f;  // 128^-0.5
  size_t src = (size_t)bs * EE + h * DD + j;
  float q1 = qf[src], q2 = qf[src + 64];
  float k1 = kf[src] * SC, k2 = kf[src + 64] * SC;
  size_t dst = (size_t)bh * SS * DD + (size_t)s * DD + j;
  qbh[dst]      = f2bf(q1 * cs - q2 * sn);
  qbh[dst + 64] = f2bf(q1 * sn + q2 * cs);
  kbh[dst]      = f2bf(k1 * cs - k2 * sn);
  kbh[dst + 64] = f2bf(k1 * sn + k2 * cs);
  size_t vb = (size_t)bh * DD * SS;
  vT[vb + (size_t)j * SS + s]        = f2bf(vf[src]);
  vT[vb + (size_t)(j + 64) * SS + s] = f2bf(vf[src + 64]);
}

// ---------------- Retention core: ret = (q kT * D) v + (q hstate) * crossdecay ----------------

__global__ __launch_bounds__(32) void k_ret(const u16* __restrict__ qbh, const u16* __restrict__ kbh,
                                            const u16* __restrict__ vT, const u16* __restrict__ hsT,
                                            const int* __restrict__ cseg, float* __restrict__ ret){
  int lane = threadIdx.x;
  int nt = blockIdx.x;               // 16-row tile
  int bh = blockIdx.y;
  int b = bh >> 3, h = bh & 7;
  int n0 = nt * 16;
  const u16* q  = qbh + (size_t)bh * SS * DD;
  const u16* k  = kbh + (size_t)bh * SS * DD;
  const u16* v  = vT  + (size_t)bh * DD * SS;
  const u16* hm = hsT + (size_t)bh * DD * DD;
  float kappa = 1.f - exp2f(-5.f - (float)h);
  float l2k = log2f(kappa);
  int hi = (lane < 16) ? 0 : 8;
  int ml = lane & 15;
  int nr[8]; int cn[8]; float crossf[8];
  #pragma unroll
  for (int r = 0; r < 8; ++r){
    nr[r] = n0 + r + hi;
    cn[r] = cseg[b * SS + nr[r]];
    crossf[r] = (cn[r] == 0) ? exp2f(l2k * (float)(nr[r] + 1)) : 0.f;
  }
  bf16x16 aq[4];
  #pragma unroll
  for (int kk = 0; kk < 4; ++kk) aq[kk] = load_a16(q, DD, n0, kk * 32, lane);
  floatx8 zero = {0.f,0.f,0.f,0.f,0.f,0.f,0.f,0.f};
  floatx8 acc[8];
  #pragma unroll
  for (int t = 0; t < 8; ++t) acc[t] = zero;
  // cross term: q @ hstate (hsT is (e,dd) row-major -> contiguous-K B frags)
  #pragma unroll
  for (int t = 0; t < 8; ++t)
    #pragma unroll
    for (int kk = 0; kk < 4; ++kk)
      acc[t] = wmma_bf(aq[kk], load_b16(hm, DD, t * 16, kk * 32, lane), acc[t]);
  #pragma unroll
  for (int t = 0; t < 8; ++t)
    #pragma unroll
    for (int r = 0; r < 8; ++r) acc[t][r] *= crossf[r];

  __shared__ alignas(16) u16 smat[16 * 32];
  for (int mt = 0; mt <= nt; mt += 2){
    int m0a = mt * 16;
    bool has2 = (mt + 1) <= nt;
    floatx8 s0 = zero, s1 = zero;
    #pragma unroll
    for (int kk = 0; kk < 4; ++kk) s0 = wmma_bf(aq[kk], load_b16(k, DD, m0a, kk * 32, lane), s0);
    if (has2){
      #pragma unroll
      for (int kk = 0; kk < 4; ++kk) s1 = wmma_bf(aq[kk], load_b16(k, DD, m0a + 16, kk * 32, lane), s1);
    }
    int m0i = m0a + ml;
    int m1i = m0i + 16;
    int cm0 = cseg[b * SS + m0i];
    int cm1 = has2 ? cseg[b * SS + m1i] : 0;
    #pragma unroll
    for (int r = 0; r < 8; ++r){
      float f0 = (nr[r] >= m0i && cn[r] == cm0) ? exp2f(l2k * (float)(nr[r] - m0i)) : 0.f;
      s0[r] *= f0;
      float f1 = (has2 && nr[r] >= m1i && cn[r] == cm1) ? exp2f(l2k * (float)(nr[r] - m1i)) : 0.f;
      s1[r] *= f1;
    }
    __syncthreads();
    #pragma unroll
    for (int r = 0; r < 8; ++r){
      smat[(r + hi) * 32 + ml]      = f2bf(s0[r]);
      smat[(r + hi) * 32 + 16 + ml] = f2bf(s1[r]);
    }
    __syncthreads();
    bf16x16 as = load_a16(smat, 32, 0, 0, lane);
    #pragma unroll
    for (int t = 0; t < 8; ++t)
      acc[t] = wmma_bf(as, load_b16(v, SS, t * 16, m0a, lane), acc[t]);
  }
  #pragma unroll
  for (int t = 0; t < 8; ++t)
    #pragma unroll
    for (int r = 0; r < 8; ++r)
      ret[(size_t)(b * SS + nr[r]) * EE + h * DD + t * 16 + ml] = acc[t][r];
}

// ---------------- state update: hnew = hs * kappa^S * keep + sum_m k[m]*decay ⊗ v[m] ----------------

__global__ __launch_bounds__(256) void k_kv(const u16* __restrict__ kbh, const u16* __restrict__ vT,
                                            const float* __restrict__ hs_in, const int* __restrict__ cseg,
                                            float* __restrict__ hs_out){
  int bh = blockIdx.x;
  int b = bh >> 3, h = bh & 7;
  int t = threadIdx.x;
  int dd = t >> 1, e0 = (t & 1) * 64;
  float kappa = 1.f - exp2f(-5.f - (float)h);
  float l2k = log2f(kappa);
  int cl = cseg[b * SS + SS - 1];
  const u16* k = kbh + (size_t)bh * SS * DD;
  const u16* v = vT  + (size_t)bh * DD * SS;
  float acc[64];
  #pragma unroll
  for (int e = 0; e < 64; ++e) acc[e] = 0.f;
  for (int m = 0; m < SS; ++m){
    if (cseg[b * SS + m] != cl) continue;
    float w = exp2f(l2k * (float)(SS - 1 - m));
    float kv = bf2f(k[(size_t)m * DD + dd]) * w;
    #pragma unroll
    for (int e = 0; e < 64; ++e)
      acc[e] += kv * bf2f(v[(size_t)(e0 + e) * SS + m]);
  }
  float keep = (cl == 0) ? exp2f(l2k * (float)SS) : 0.f;
  size_t base = (size_t)bh * DD * DD + (size_t)dd * DD + e0;
  #pragma unroll
  for (int e = 0; e < 64; ++e)
    hs_out[base + e] = hs_in[base + e] * keep + acc[e];
}

// ---------------- per-head groupnorm * gscale ----------------

__global__ __launch_bounds__(128) void k_gnorm(const float* __restrict__ ret, const float* __restrict__ g,
                                               float* __restrict__ out){
  int bsh = blockIdx.x;
  int bs = bsh >> 3, h = bsh & 7;
  int j = threadIdx.x;
  float x = ret[(size_t)bs * EE + h * DD + j];
  __shared__ float s1[128], s2[128];
  s1[j] = x; s2[j] = x * x;
  __syncthreads();
  for (int off = 64; off > 0; off >>= 1){
    if (j < off){ s1[j] += s1[j + off]; s2[j] += s2[j + off]; }
    __syncthreads();
  }
  float mu = s1[0] * (1.f / 128.f);
  float var = s2[0] * (1.f / 128.f) - mu * mu;
  float rstd = rsqrtf(var + 1e-6f);
  out[(size_t)bs * EE + h * DD + j] = (x - mu) * rstd * g[h * DD + j];
}

// ---------------- silu(g) * r, write bf16 or f32 ----------------

__global__ void k_silumul(const float* __restrict__ gf, const float* __restrict__ rf,
                          u16* __restrict__ out_bf, float* __restrict__ out_f, size_t n){
  size_t i = (size_t)blockIdx.x * blockDim.x + threadIdx.x;
  if (i >= n) return;
  float g = gf[i];
  float val = (g / (1.f + __expf(-g))) * rf[i];
  if (out_bf) out_bf[i] = f2bf(val);
  else        out_f[i] = val;
}

// ---------------- rmsnorm(a + b) * w, optional f32/bf16 outputs ----------------

__global__ __launch_bounds__(256) void k_addrms(const float* __restrict__ a, const float* __restrict__ b,
                                                const float* __restrict__ w,
                                                float* __restrict__ out_f, u16* __restrict__ out_bf){
  int row = blockIdx.x;
  int t = threadIdx.x;
  float x[4]; float ss = 0.f;
  #pragma unroll
  for (int j = 0; j < 4; ++j){
    int col = t + j * 256;
    x[j] = a[(size_t)row * EE + col] + b[(size_t)row * EE + col];
    ss += x[j] * x[j];
  }
  __shared__ float sred[256];
  sred[t] = ss;
  __syncthreads();
  for (int off = 128; off > 0; off >>= 1){
    if (t < off) sred[t] += sred[t + off];
    __syncthreads();
  }
  float rms = rsqrtf(sred[0] * (1.f / (float)EE) + 1e-6f);
  #pragma unroll
  for (int j = 0; j < 4; ++j){
    int col = t + j * 256;
    float y = x[j] * rms * w[col];
    if (out_f)  out_f[(size_t)row * EE + col] = y;
    if (out_bf) out_bf[(size_t)row * EE + col] = f2bf(y);
  }
}

// ---------------- host-side orchestration ----------------

static void msr_pass(const u16* qin_bf, const u16* kin_bf,
                     const u16* hsT, const float* hs_in,
                     const u16* wqT, const u16* wkT, const u16* wvTp, const u16* wgT, const u16* woT,
                     const float* g, const int* pos, const int* cseg,
                     float* bufA, float* bufB, float* bufC,
                     u16* qbh, u16* kbh, u16* vbhT, u16* smbf,
                     float* hs_out, hipStream_t stream){
  dim3 gg(EE / 128, MM / 128);
  k_gemm<<<gg, 256, 0, stream>>>(qin_bf, wqT, bufA);
  k_gemm<<<gg, 256, 0, stream>>>(kin_bf, wkT, bufB);
  k_gemm<<<gg, 256, 0, stream>>>(kin_bf, wvTp, bufC);
  k_qkvpost<<<MM, 512, 0, stream>>>(bufA, bufB, bufC, pos, qbh, kbh, vbhT);
  k_ret<<<dim3(SS / 16, BB * HH), 32, 0, stream>>>(qbh, kbh, vbhT, hsT, cseg, bufA);
  k_kv<<<BB * HH, 256, 0, stream>>>(kbh, vbhT, hs_in, cseg, hs_out);
  k_gnorm<<<MM * HH, 128, 0, stream>>>(bufA, g, bufC);
  k_gemm<<<gg, 256, 0, stream>>>(qin_bf, wgT, bufB);
  k_silumul<<<(MM * EE) / 256, 256, 0, stream>>>(bufB, bufC, smbf, (float*)nullptr, (size_t)MM * EE);
  k_gemm<<<gg, 256, 0, stream>>>(smbf, woT, bufA);
}

extern "C" void kernel_launch(void* const* d_in, const int* in_sizes, int n_in,
                              void* d_out, int out_size, void* d_ws, size_t ws_size,
                              hipStream_t stream){
  (void)in_sizes; (void)n_in; (void)out_size; (void)ws_size;
  const float* x    = (const float*)d_in[0];
  const float* obs  = (const float*)d_in[1];
  const float* hs1  = (const float*)d_in[2];
  const float* hs2  = (const float*)d_in[3];
  const float* W[12] = {
    (const float*)d_in[4],  (const float*)d_in[5],  (const float*)d_in[6],
    (const float*)d_in[7],  (const float*)d_in[8],
    (const float*)d_in[10], (const float*)d_in[11], (const float*)d_in[12],
    (const float*)d_in[13], (const float*)d_in[14],
    (const float*)d_in[19], (const float*)d_in[20]
  };
  const float* g1   = (const float*)d_in[9];
  const float* g2   = (const float*)d_in[15];
  const float* ln1w = (const float*)d_in[16];
  const float* ln2w = (const float*)d_in[17];
  const float* ln3w = (const float*)d_in[18];
  const unsigned char* dones = (const unsigned char*)d_in[21];
  const int* pos = (const int*)d_in[22];

  float* outp = (float*)d_out;
  float* hs1n = outp + (size_t)MM * EE;
  float* hs2n = hs1n + (size_t)BB * HH * DD * DD;

  // workspace carve-up (256B-aligned slots)
  char* ws = (char*)d_ws;
  size_t off = 0;
  auto carve = [&](size_t bytes) -> void* {
    void* p = ws + off;
    off = (off + bytes + 255) & ~(size_t)255;
    return p;
  };
  u16* wT[12];
  for (int i = 0; i < 12; ++i) wT[i] = (u16*)carve((size_t)EE * EE * 2);
  u16* xbf   = (u16*)carve((size_t)MM * EE * 2);
  u16* obsbf = (u16*)carve((size_t)MM * EE * 2);
  u16* t1bf  = (u16*)carve((size_t)MM * EE * 2);
  u16* qbh   = (u16*)carve((size_t)BB * HH * SS * DD * 2 + 4096);
  u16* kbh   = (u16*)carve((size_t)BB * HH * SS * DD * 2 + 4096);
  u16* vbhT  = (u16*)carve((size_t)BB * HH * DD * SS * 2 + 4096);
  u16* hsT1  = (u16*)carve((size_t)BB * HH * DD * DD * 2);
  u16* hsT2  = (u16*)carve((size_t)BB * HH * DD * DD * 2);
  int* cseg  = (int*)carve((size_t)BB * SS * 4);
  float* bufA = (float*)carve((size_t)MM * EE * 4);
  float* bufB = (float*)carve((size_t)MM * EE * 4);
  float* bufC = (float*)carve((size_t)MM * EE * 4);

  size_t nact = (size_t)MM * EE;
  // one-time conversions
  k_cvt<<<(unsigned)((nact + 255) / 256), 256, 0, stream>>>(x, xbf, nact);
  k_cvt<<<(unsigned)((nact + 255) / 256), 256, 0, stream>>>(obs, obsbf, nact);
  for (int i = 0; i < 12; ++i)
    k_trans<<<dim3(EE / 16, EE / 16, 1), dim3(16, 16), 0, stream>>>(W[i], wT[i], EE);
  k_trans<<<dim3(DD / 16, DD / 16, BB * HH), dim3(16, 16), 0, stream>>>(hs1, hsT1, DD);
  k_trans<<<dim3(DD / 16, DD / 16, BB * HH), dim3(16, 16), 0, stream>>>(hs2, hsT2, DD);
  k_seg<<<1, 8, 0, stream>>>(dones, cseg);

  // layer 1: r1 = msr(x, x, x, hs1)
  msr_pass(xbf, xbf, hsT1, hs1, wT[0], wT[1], wT[2], wT[3], wT[4], g1, pos, cseg,
           bufA, bufB, bufC, qbh, kbh, vbhT, kbh, hs1n, stream);
  // t1 = rmsnorm(x + r1, ln1)  (bf16 only; GEMM input for layer 2)
  k_addrms<<<MM, 256, 0, stream>>>(x, bufA, ln1w, (float*)nullptr, t1bf);

  // layer 2: r2 = msr(obs_rep, t1, t1, hs2)
  msr_pass(obsbf, t1bf, hsT2, hs2, wT[5], wT[6], wT[7], wT[8], wT[9], g2, pos, cseg,
           bufA, bufB, bufC, qbh, kbh, vbhT, kbh, hs2n, stream);
  // y = rmsnorm(obs_rep + r2, ln2) -> f32 (bufB) and bf16 (qbh)
  k_addrms<<<MM, 256, 0, stream>>>(obs, bufA, ln2w, bufB, qbh);

  // FFN: silu(y @ gate) * (y @ proj); out = rmsnorm(y + ffn, ln3)
  dim3 gg(EE / 128, MM / 128);
  k_gemm<<<gg, 256, 0, stream>>>(qbh, wT[10], bufC);
  k_gemm<<<gg, 256, 0, stream>>>(qbh, wT[11], bufA);
  k_silumul<<<(MM * EE) / 256, 256, 0, stream>>>(bufC, bufA, (u16*)nullptr, bufC, nact);
  k_addrms<<<MM, 256, 0, stream>>>(bufB, bufC, ln3w, outp, (u16*)nullptr);
}
